// VectorQuantizer_15290083574033
// MI455X (gfx1250) — matrix operand order repaired and checked by reading, not compiled, not run on hardware
//
#include <hip/hip_runtime.h>
#include <stdint.h>

#define Nv 131072
#define Kv 4096
#define Dv 64
#define BETAv 0.25f
#define DECAYv 0.99f
#define EPSv 1e-5f

typedef __attribute__((ext_vector_type(16))) __bf16 v16bf;
typedef __attribute__((ext_vector_type(8)))  float  v8f;

__device__ __forceinline__ unsigned short f2bf_bits(float f) {
    union { float f; unsigned u; } a; a.f = f;
    unsigned r = (a.u + 0x7FFFu + ((a.u >> 16) & 1u)) >> 16;  // RNE
    return (unsigned short)r;
}
__device__ __forceinline__ __bf16 f2bf16(float f) {
    unsigned short s = f2bf_bits(f);
    __bf16 b;
    __builtin_memcpy(&b, &s, 2);
    return b;
}

// ---------------- Kernel 1: prep (bf16 codebook, norms, zero accumulators) ----------------
__global__ void vq_prep(const float* __restrict__ cb, unsigned short* __restrict__ cbh,
                        float* __restrict__ cbnorm, float* __restrict__ hist,
                        float* __restrict__ emb, float* __restrict__ scalars) {
    int k = blockIdx.x;
    int d = threadIdx.x;                  // 64 threads
    float v = cb[k * Dv + d];
    cbh[k * Dv + d] = f2bf_bits(v);
    emb[k * Dv + d] = 0.0f;
    __shared__ float red[64];
    red[d] = v * v;
    __syncthreads();
    if (d == 0) {
        float s = 0.0f;
#pragma unroll
        for (int i = 0; i < 64; ++i) s += red[i];
        cbnorm[k] = s;
        hist[k] = 0.0f;
    }
    if (k == 0 && d < 2) scalars[d] = 0.0f;   // [0]=loss accum, [1]=n accum
}

// load one 16-row x 64-k bf16 A-fragment pair for a z tile (ISA 16-bit A layout)
__device__ __forceinline__ void load_a_frags(const float* zr, bool hi, v16bf& a0, v16bf& a1) {
    const float4* z4 = (const float4*)zr;
    const int klo = hi ? 8 : 0;
    float4 p[8];
#pragma unroll
    for (int s = 0; s < 4; ++s) {
        p[2 * s]     = z4[(klo + 16 * s) / 4];
        p[2 * s + 1] = z4[(klo + 16 * s) / 4 + 1];
    }
#pragma unroll
    for (int s = 0; s < 2; ++s) {
        const float4 q0 = p[2 * s], q1 = p[2 * s + 1];
        const float4 q2 = p[2 * s + 4], q3 = p[2 * s + 5];
        a0[8 * s + 0] = f2bf16(q0.x); a0[8 * s + 1] = f2bf16(q0.y);
        a0[8 * s + 2] = f2bf16(q0.z); a0[8 * s + 3] = f2bf16(q0.w);
        a0[8 * s + 4] = f2bf16(q1.x); a0[8 * s + 5] = f2bf16(q1.y);
        a0[8 * s + 6] = f2bf16(q1.z); a0[8 * s + 7] = f2bf16(q1.w);
        a1[8 * s + 0] = f2bf16(q2.x); a1[8 * s + 1] = f2bf16(q2.y);
        a1[8 * s + 2] = f2bf16(q2.z); a1[8 * s + 3] = f2bf16(q2.w);
        a1[8 * s + 4] = f2bf16(q3.x); a1[8 * s + 5] = f2bf16(q3.y);
        a1[8 * s + 6] = f2bf16(q3.z); a1[8 * s + 7] = f2bf16(q3.w);
    }
}

// ---------------- Kernel 2: WMMA distance + argmin ----------------
// grid = N/256 blocks, 256 threads (8 waves); wave owns 32 rows of z (two A tiles)
// so each B fragment pair feeds 4 WMMAs (WMMA:DS ratio 1:1).
__global__ __launch_bounds__(256) void vq_argmin(
    const float* __restrict__ z, const unsigned short* __restrict__ cbh,
    const float* __restrict__ cbnorm, float* __restrict__ idx_out) {

    const int tid   = threadIdx.x;
    const int lane  = tid & 31;
    const int wave  = tid >> 5;
    const int lhalf = lane & 15;
    const bool hi   = lane >= 16;

    // 128 entries/chunk, row padded 128B -> 144B (72 ushorts) to kill LDS bank conflicts
    __shared__ __align__(16) unsigned short cbS[2][128 * 72];

    const int rowbase = blockIdx.x * 256 + wave * 32;
    v16bf a0, a1, a2, a3;
    load_a_frags(z + (size_t)(rowbase + lhalf)      * Dv, hi, a0, a1);  // rows 0..15
    load_a_frags(z + (size_t)(rowbase + 16 + lhalf) * Dv, hi, a2, a3);  // rows 16..31

    const uint64_t gbase = (uint64_t)(uintptr_t)cbh;
    const unsigned lb0 = (unsigned)(uintptr_t)(&cbS[0][0]);
    const unsigned lb1 = (unsigned)(uintptr_t)(&cbS[1][0]);

    // async prefetch of one 128-entry chunk (16 KB) into LDS, padded layout
    auto prefetch = [&](int buf, int chunk) {
        const unsigned lb = buf ? lb1 : lb0;
        const uint64_t gb = gbase + (uint64_t)chunk * (128 * 64 * 2);
#pragma unroll
        for (int i = 0; i < 4; ++i) {
            unsigned u   = (unsigned)tid * 4u + (unsigned)i;    // 16B-unit index 0..1023
            unsigned lds = lb + ((u + (u >> 3)) << 4);          // +16B pad per entry
            unsigned gv  = u << 4;
            asm volatile("global_load_async_to_lds_b128 %0, %1, %2"
                         :: "v"(lds), "v"(gv), "s"(gb) : "memory");
        }
    };

    float    best0[8], best1[8];
    unsigned bidx0[8], bidx1[8];
#pragma unroll
    for (int r = 0; r < 8; ++r) {
        best0[r] = 3.4e38f; bidx0[r] = 0u;
        best1[r] = 3.4e38f; bidx1[r] = 0u;
    }

    prefetch(0, 0);

    for (int c = 0; c < 32; ++c) {
        if (c + 1 < 32) {
            prefetch((c + 1) & 1, c + 1);
            asm volatile("s_wait_asynccnt 4" ::: "memory");   // chunk c complete (in-order)
        } else {
            asm volatile("s_wait_asynccnt 0" ::: "memory");
        }
        __syncthreads();
        const unsigned short* buf = &cbS[c & 1][0];

#pragma unroll 4
        for (int t = 0; t < 8; ++t) {
            const int eloc = t * 16 + lhalf;                        // B column = codebook entry
            const unsigned short* rowp = buf + eloc * 72 + (hi ? 8 : 0);
            union { uint4 q[2]; v16bf v; } b0u, b1u;
            const uint4* bp = (const uint4*)rowp;
            b0u.q[0] = bp[0]; b0u.q[1] = bp[2];                     // k 0..31 half
            b1u.q[0] = bp[4]; b1u.q[1] = bp[6];                     // k 32..63 half

            const int n0 = c * 128 + t * 16 + lhalf;
            const float nrm = cbnorm[n0];

            v8f accA = {};
            accA = __builtin_amdgcn_wmma_f32_16x16x32_bf16(false, a0, false, b0u.v,
                                                           (short)0, accA, false, false);
            accA = __builtin_amdgcn_wmma_f32_16x16x32_bf16(false, a1, false, b1u.v,
                                                           (short)0, accA, false, false);
            v8f accB = {};
            accB = __builtin_amdgcn_wmma_f32_16x16x32_bf16(false, a2, false, b0u.v,
                                                           (short)0, accB, false, false);
            accB = __builtin_amdgcn_wmma_f32_16x16x32_bf16(false, a3, false, b1u.v,
                                                           (short)0, accB, false, false);
#pragma unroll
            for (int r = 0; r < 8; ++r) {
                float vA = fmaf(-2.0f, accA[r], nrm);   // ||e||^2 - 2 z.e  (||z||^2 const)
                if (vA < best0[r]) { best0[r] = vA; bidx0[r] = (unsigned)n0; }
                float vB = fmaf(-2.0f, accB[r], nrm);
                if (vB < best1[r]) { best1[r] = vB; bidx1[r] = (unsigned)n0; }
            }
        }
        __syncthreads();   // buf (c&1) free for reuse by chunk c+2
    }

    // reduce min+index across the 16 lanes that hold the same row (wave32 shuffles)
#pragma unroll
    for (int r = 0; r < 8; ++r) {
        float b = best0[r]; unsigned i = bidx0[r];
        float c2 = best1[r]; unsigned j = bidx1[r];
#pragma unroll
        for (int m = 8; m >= 1; m >>= 1) {
            float    ob = __shfl_xor(b, m, 32);
            unsigned oi = __shfl_xor(i, m, 32);
            if (ob < b || (ob == b && oi < i)) { b = ob; i = oi; }
            float    oc = __shfl_xor(c2, m, 32);
            unsigned oj = __shfl_xor(j, m, 32);
            if (oc < c2 || (oc == c2 && oj < j)) { c2 = oc; j = oj; }
        }
        bidx0[r] = i; bidx1[r] = j;
    }
    if (lane == 0) {
#pragma unroll
        for (int r = 0; r < 8; ++r) {
            idx_out[rowbase + r]      = (float)bidx0[r];
            idx_out[rowbase + 16 + r] = (float)bidx1[r];
        }
    } else if (lane == 16) {
#pragma unroll
        for (int r = 0; r < 8; ++r) {
            idx_out[rowbase + 8 + r]  = (float)bidx0[r];
            idx_out[rowbase + 24 + r] = (float)bidx1[r];
        }
    }
}

// ---------------- Kernel 3: gather quantized (exact fp32), loss, histogram, segment-sum ----------------
__global__ __launch_bounds__(256) void vq_scatter(
    const float* __restrict__ z, const float* __restrict__ cb,
    const float* __restrict__ idx_in, float* __restrict__ qout,
    float* __restrict__ hist, float* __restrict__ emb, float* __restrict__ scalars) {
    int tid = threadIdx.x;
    int r = tid >> 6, d = tid & 63;
    int row = blockIdx.x * 4 + r;
    int idx = (int)idx_in[row];
    float zv = z[(size_t)row * Dv + d];
    float qv = cb[(size_t)idx * Dv + d];
    qout[(size_t)row * Dv + d] = qv;
    float diff = zv - qv;
    atomicAdd(&emb[(size_t)idx * Dv + d], zv);
    if (d == 0) atomicAdd(&hist[idx], 1.0f);
    __shared__ float red[256];
    red[tid] = diff * diff;
    __syncthreads();
    for (int s = 128; s > 0; s >>= 1) {
        if (tid < s) red[tid] += red[tid + s];
        __syncthreads();
    }
    if (tid == 0) atomicAdd(&scalars[0], red[0]);
}

// ---------------- Kernel 4: EMA blend + sum(new_cluster_size) ----------------
__global__ void vq_ema(const float* __restrict__ ema_cs, const float* __restrict__ ema_es,
                       const float* __restrict__ hist, const float* __restrict__ emb,
                       float* __restrict__ ncs_out, float* __restrict__ nes_out,
                       float* __restrict__ scalars) {
    int k = blockIdx.x, d = threadIdx.x;
    nes_out[k * Dv + d] = DECAYv * ema_es[k * Dv + d] + (1.0f - DECAYv) * emb[k * Dv + d];
    if (d == 0) {
        float v = DECAYv * ema_cs[k] + (1.0f - DECAYv) * hist[k];
        ncs_out[k] = v;
        atomicAdd(&scalars[1], v);
    }
}

// ---------------- Kernel 5: Laplace-smoothed normalize + loss scalar ----------------
__global__ void vq_final(const float* __restrict__ ncs, const float* __restrict__ nes,
                         float* __restrict__ ncb_out, float* __restrict__ loss_out,
                         const float* __restrict__ scalars) {
    int k = blockIdx.x, d = threadIdx.x;
    float n = scalars[1];
    float sm = (ncs[k] + EPSv) / (n + (float)Kv * EPSv) * n;
    ncb_out[k * Dv + d] = nes[k * Dv + d] / sm;
    if (k == 0 && d == 0) loss_out[0] = BETAv * scalars[0] / ((float)Nv * (float)Dv);
}

extern "C" void kernel_launch(void* const* d_in, const int* in_sizes, int n_in,
                              void* d_out, int out_size, void* d_ws, size_t ws_size,
                              hipStream_t stream) {
    (void)in_sizes; (void)n_in; (void)out_size; (void)ws_size;
    const float* z   = (const float*)d_in[0];
    const float* cb  = (const float*)d_in[1];
    const float* ecs = (const float*)d_in[2];
    const float* ees = (const float*)d_in[3];
    float* out = (float*)d_out;

    char* ws = (char*)d_ws;
    unsigned short* cbh  = (unsigned short*)ws;            // K*D*2  = 512 KB
    float* cbnorm = (float*)(ws + 524288);                 // K*4    = 16 KB
    float* hist   = (float*)(ws + 540672);                 // K*4    = 16 KB
    float* emb    = (float*)(ws + 557056);                 // K*D*4  = 1 MB
    float* scalars= (float*)(ws + 1605632);                // 2 floats

    // output layout (flat float, reference tuple order)
    float* q_out    = out;                                 // N*D
    float* loss_out = out + (size_t)Nv * Dv;               // 1
    float* idx_out  = loss_out + 1;                        // N
    float* ncs_out  = idx_out + Nv;                        // K
    float* nes_out  = ncs_out + Kv;                        // K*D
    float* ncb_out  = nes_out + (size_t)Kv * Dv;           // K*D

    vq_prep  <<<Kv,       64, 0, stream>>>(cb, cbh, cbnorm, hist, emb, scalars);
    vq_argmin<<<Nv / 256, 256, 0, stream>>>(z, cbh, cbnorm, idx_out);
    vq_scatter<<<Nv / 4,  256, 0, stream>>>(z, cb, idx_out, q_out, hist, emb, scalars);
    vq_ema   <<<Kv,       64, 0, stream>>>(ecs, ees, hist, emb, ncs_out, nes_out, scalars);
    vq_final <<<Kv,       64, 0, stream>>>(ncs_out, nes_out, ncb_out, loss_out, scalars);
}
